// Coords2Stress_44066364457320
// MI455X (gfx1250) — matrix operand order repaired and checked by reading, not compiled, not run on hardware
//
#include <hip/hip_runtime.h>
#include <hip/hip_bf16.h>
#include <cstdint>

// ---------------------------------------------------------------------------
// Coords2Stress for MI455X (gfx1250, wave32)
//   1) build_hessian_kernel : writes H [B,3N,3N] fp32 to d_out AND (if ws is
//      large enough) a bf16 mirror to d_ws -- both from registers, one pass.
//   2) power iteration      : w = H v via v_wmma_f32_16x16x32_bf16 reading the
//      bf16 mirror (75.5 MB/sweep, resident in the 192 MB L2 -> L2-BW bound).
//      K unrolled x2 with dual accumulators so 8 b128 loads are in flight per
//      iteration and the two WMMA chains are independent. Normalize every 3
//      steps (range-safe) to halve launch count. Fallback: fp32-read +
//      in-register cvt matvec when ws is too small for the mirror.
//   3) 4 fp32 VALU matvec polish steps + fp32 Rayleigh lambda; disp = 3*l*v.
// ---------------------------------------------------------------------------

typedef __bf16 bf16_t;
typedef bf16_t v8bf  __attribute__((ext_vector_type(8)));
typedef bf16_t v16bf __attribute__((ext_vector_type(16)));
typedef float  v8f   __attribute__((ext_vector_type(8)));

#define INV_CUTOFF2 (1.0f / 49.0f)
#define EPS_D2 1e-5f

__device__ __forceinline__ float wave_reduce_sum(float x) {
  #pragma unroll
  for (int off = 16; off > 0; off >>= 1)
    x += __shfl_down(x, off, 32);
  return x;
}

// one 16x32 bf16 A-tile / 32x16 broadcast-B WMMA step at K offset kb
__device__ __forceinline__ v8f wmma_step(const bf16_t* __restrict__ hrow,
                                         const bf16_t* __restrict__ vb,
                                         int kb, int half, v8f acc) {
  const bf16_t* hr = hrow + kb + half * 16;
  const v8bf a0 = *(const v8bf*)(hr);          // A: K kb+off .. +7   (16B)
  const v8bf a1 = *(const v8bf*)(hr + 8);      // A: K kb+off+8 .. +15
  const v16bf a = __builtin_shufflevector(a0, a1, 0, 1, 2, 3, 4, 5, 6, 7,
                                          8, 9, 10, 11, 12, 13, 14, 15);
  const v16bf bv = *(const v16bf*)(vb + kb + half * 16);  // B: v broadcast
  return __builtin_amdgcn_wmma_f32_16x16x32_bf16(
      false, a, false, bv, (short)0, acc, false, false);
}

// --------------------------- Hessian build ---------------------------------
// grid = (N, B), block = 256. Block owns atom i of batch b: writes the three
// H rows 3i..3i+2 (zeros outside the valid block), the reduced diagonal 3x3
// block, and (optionally) the bf16 mirror of everything it writes.
__global__ __launch_bounds__(256) void build_hessian_kernel(
    const float* __restrict__ coords, const int* __restrict__ num_atoms,
    float* __restrict__ H, bf16_t* __restrict__ H16 /* nullable */,
    int N, int M /* = 3N */) {
  const int i = blockIdx.x;
  const int b = blockIdx.y;
  const int tid = threadIdx.x;
  const int m = num_atoms[b];
  const float* cb = coords + (size_t)b * M;
  float* Hb = H + (size_t)b * M * M;
  bf16_t* Hb16 = H16 ? H16 + (size_t)b * M * M : nullptr;
  const bool vi = (i < m);
  const float xi = cb[3 * i + 0], yi = cb[3 * i + 1], zi = cb[3 * i + 2];

  float d00 = 0.f, d01 = 0.f, d02 = 0.f, d11 = 0.f, d12 = 0.f, d22 = 0.f;

  for (int j = tid; j < N; j += 256) {
    float h00 = 0.f, h01 = 0.f, h02 = 0.f, h11 = 0.f, h12 = 0.f, h22 = 0.f;
    if (vi && j < m) {
      const float sx = xi - cb[3 * j + 0];
      const float sy = yi - cb[3 * j + 1];
      const float sz = zi - cb[3 * j + 2];
      const float d2 = sx * sx + sy * sy + sz * sz + EPS_D2;
      const float g = __expf(-d2 * INV_CUTOFF2) / d2;   // i==j -> sep=0 -> h=0
      h00 = -sx * sx * g; h01 = -sx * sy * g; h02 = -sx * sz * g;
      h11 = -sy * sy * g; h12 = -sy * sz * g; h22 = -sz * sz * g;
      d00 -= h00; d01 -= h01; d02 -= h02; d11 -= h11; d12 -= h12; d22 -= h22;
    }
    size_t r = (size_t)(3 * i + 0) * M + 3 * j;
    Hb[r + 0] = h00; Hb[r + 1] = h01; Hb[r + 2] = h02;
    Hb[r + M + 0] = h01; Hb[r + M + 1] = h11; Hb[r + M + 2] = h12;
    Hb[r + 2 * (size_t)M + 0] = h02; Hb[r + 2 * (size_t)M + 1] = h12;
    Hb[r + 2 * (size_t)M + 2] = h22;
    if (Hb16) {
      Hb16[r + 0] = (bf16_t)h00; Hb16[r + 1] = (bf16_t)h01; Hb16[r + 2] = (bf16_t)h02;
      Hb16[r + M + 0] = (bf16_t)h01; Hb16[r + M + 1] = (bf16_t)h11; Hb16[r + M + 2] = (bf16_t)h12;
      Hb16[r + 2 * (size_t)M + 0] = (bf16_t)h02; Hb16[r + 2 * (size_t)M + 1] = (bf16_t)h12;
      Hb16[r + 2 * (size_t)M + 2] = (bf16_t)h22;
    }
  }

  // reduce the 6 unique diagonal-block components across the block
  float dc[6] = {d00, d01, d02, d11, d12, d22};
  #pragma unroll
  for (int c = 0; c < 6; ++c) dc[c] = wave_reduce_sum(dc[c]);
  __shared__ float sdg[8][6];
  if ((tid & 31) == 0) {
    #pragma unroll
    for (int c = 0; c < 6; ++c) sdg[tid >> 5][c] = dc[c];
  }
  __syncthreads();
  if (tid < 6 && vi) {
    float s = 0.f;
    #pragma unroll
    for (int k = 0; k < 8; ++k) s += sdg[k][tid];
    const int P[6] = {0, 0, 0, 1, 1, 2};
    const int Q[6] = {0, 1, 2, 1, 2, 2};
    const int p = P[tid], q = Q[tid];
    const size_t e0 = (size_t)(3 * i + p) * M + 3 * i + q;
    const size_t e1 = (size_t)(3 * i + q) * M + 3 * i + p;
    Hb[e0] = s;
    if (p != q) Hb[e1] = s;
    if (Hb16) {
      Hb16[e0] = (bf16_t)s;
      if (p != q) Hb16[e1] = (bf16_t)s;
    }
  }
}

// ---------------- WMMA matvec, pure-bf16 operands (fast path) --------------
// grid = (M/16, B), block = 128 (4 waves). Wave owns a 16-row A strip and a
// quarter of K. A/B use contiguous K (k = 16*half + s), not the ISA table: A lanes<16 hold K
// 0..15 of row lane&15; B lanes<16 hold K 0..15 of column lane&15 (the
// iterate is broadcast into all 16 columns, so column 0 of D is H*v).
// K unrolled x2 with independent accumulators; writes the strip of w in fp32
// AND bf16 so matvecs chain without an intermediate convert kernel.
__global__ __launch_bounds__(128) void matvec_wmma_bf16_kernel(
    const bf16_t* __restrict__ H16, const bf16_t* __restrict__ vin16,
    float* __restrict__ wout, bf16_t* __restrict__ wout16, int M) {
  const int b = blockIdx.y;
  const int strip = blockIdx.x;
  const int tid = threadIdx.x;
  const int wave = tid >> 5;
  const int lane = tid & 31;
  const int half = lane >> 4;
  const int r = strip * 16 + (lane & 15);
  const bf16_t* hrow = H16 + (size_t)b * M * M + (size_t)r * M;
  const bf16_t* vb = vin16 + (size_t)b * M;

  v8f acc0 = {}, acc1 = {};
  const int kseg = M >> 2;
  const int k0 = wave * kseg;
  const int kmain = k0 + (kseg & ~63);
  const int k1 = k0 + kseg;
  for (int kb = k0; kb < kmain; kb += 64) {
    __builtin_prefetch(hrow + kb + 128 + half * 8, 0, 3);  // WGP-scope prefetch, 2 iters ahead
    acc0 = wmma_step(hrow, vb, kb, half, acc0);
    acc1 = wmma_step(hrow, vb, kb + 32, half, acc1);
  }
  for (int kb = kmain; kb < k1; kb += 32)                  // tail (empty when M%256==0)
    acc0 = wmma_step(hrow, vb, kb, half, acc0);
  #pragma unroll
  for (int g = 0; g < 8; ++g) acc0[g] += acc1[g];

  __shared__ float part[4][16];
  if ((lane & 15) == 0) {                  // column 0 lives in lanes 0 and 16
    const int rb = half * 8;
    #pragma unroll
    for (int g = 0; g < 8; ++g) part[wave][rb + g] = acc0[g];
  }
  __syncthreads();
  if (tid < 16) {
    const float s = part[0][tid] + part[1][tid] + part[2][tid] + part[3][tid];
    const size_t o = (size_t)b * M + strip * 16 + tid;
    wout[o] = s;
    wout16[o] = (bf16_t)s;
  }
}

// -------- WMMA matvec, fp32 H + in-register cvt (small-ws fallback) --------
__global__ __launch_bounds__(128) void matvec_wmma_cvt_kernel(
    const float* __restrict__ H, const float* __restrict__ v,
    float* __restrict__ w, int M) {
  const int b = blockIdx.y;
  const int strip = blockIdx.x;
  const int tid = threadIdx.x;
  const int wave = tid >> 5;
  const int lane = tid & 31;
  const int half = lane >> 4;
  const int r = strip * 16 + (lane & 15);
  const float* hrow = H + (size_t)b * M * M + (size_t)r * M;
  const float* vb = v + (size_t)b * M;

  v8f acc = {};
  const int kseg = M >> 2;
  const int k0 = wave * kseg;
  const int k1 = k0 + kseg;
  for (int kb = k0; kb < k1; kb += 32) {
    const float* hr = hrow + kb + half * 16;
    const float* vs = vb + kb + half * 16;
    v16bf a, bv;
    #pragma unroll
    for (int t = 0; t < 8; ++t) {
      a[t]     = (bf16_t)hr[t];
      a[t + 8] = (bf16_t)hr[8 + t];
    }
    #pragma unroll
    for (int t = 0; t < 16; ++t) bv[t] = (bf16_t)vs[t];
    acc = __builtin_amdgcn_wmma_f32_16x16x32_bf16(
        false, a, false, bv, (short)0, acc, false, false);
  }

  __shared__ float part[4][16];
  if ((lane & 15) == 0) {
    const int rb = half * 8;
    #pragma unroll
    for (int g = 0; g < 8; ++g) part[wave][rb + g] = acc[g];
  }
  __syncthreads();
  if (tid < 16)
    w[(size_t)b * M + strip * 16 + tid] =
        part[0][tid] + part[1][tid] + part[2][tid] + part[3][tid];
}

// --------------------------- fp32 polish matvec ----------------------------
__global__ __launch_bounds__(256) void matvec_f32_kernel(
    const float* __restrict__ H, const float* __restrict__ v,
    float* __restrict__ w, int M) {
  const int b = blockIdx.y;
  const int tid = threadIdx.x;
  const int wave = tid >> 5, lane = tid & 31;
  const int row = blockIdx.x * 8 + wave;
  const float* Hb = H + (size_t)b * M * M + (size_t)row * M;
  const float* vb = v + (size_t)b * M;
  float s = 0.f;
  for (int k = lane; k < M; k += 32) s = fmaf(Hb[k], vb[k], s);
  s = wave_reduce_sum(s);
  if (lane == 0) w[(size_t)b * M + row] = s;
}

// ------------------- normalize + Rayleigh quotient -------------------------
// lambda = v.(Hv)/||v||^2 with v the pre-matvec iterate; v <- w/||w|| (+bf16).
__global__ __launch_bounds__(256) void normalize_kernel(
    float* __restrict__ v, const float* __restrict__ w,
    float* __restrict__ lam, bf16_t* __restrict__ v16 /* nullable */, int M) {
  const int b = blockIdx.x;
  const int tid = threadIdx.x;
  float* vb = v + (size_t)b * M;
  const float* wb = w + (size_t)b * M;
  float dot = 0.f, n2 = 0.f, vn2 = 0.f;
  for (int i = tid; i < M; i += 256) {
    const float wi = wb[i], vi = vb[i];
    dot = fmaf(vi, wi, dot);
    n2 = fmaf(wi, wi, n2);
    vn2 = fmaf(vi, vi, vn2);
  }
  dot = wave_reduce_sum(dot);
  n2 = wave_reduce_sum(n2);
  vn2 = wave_reduce_sum(vn2);
  __shared__ float sd[8], sn[8], sv[8];
  __shared__ float binv;
  if ((tid & 31) == 0) { sd[tid >> 5] = dot; sn[tid >> 5] = n2; sv[tid >> 5] = vn2; }
  __syncthreads();
  if (tid == 0) {
    float D = 0.f, Nn = 0.f, Vv = 0.f;
    #pragma unroll
    for (int k = 0; k < 8; ++k) { D += sd[k]; Nn += sn[k]; Vv += sv[k]; }
    lam[b] = D / fmaxf(Vv, 1e-30f);      // Rayleigh quotient (v not always unit)
    binv = rsqrtf(fmaxf(Nn, 1e-30f));
  }
  __syncthreads();
  const float inv = binv;
  for (int i = tid; i < M; i += 256) {
    const float nv = wb[i] * inv;
    vb[i] = nv;
    if (v16) v16[(size_t)b * M + i] = (bf16_t)nv;
  }
}

// --------------------------- init / finalize -------------------------------
__global__ void init_v_kernel(float* __restrict__ v,
                              bf16_t* __restrict__ v16 /* nullable */, int total) {
  const int i = blockIdx.x * 256 + threadIdx.x;
  if (i < total) {
    uint32_t h = (uint32_t)i * 2654435761u;
    h ^= h >> 16;
    const float x = 0.5f + (float)(h & 0xFFFFu) * (1.0f / 65536.0f);
    v[i] = x;
    if (v16) v16[i] = (bf16_t)x;
  }
}

__global__ void finalize_kernel(const float* __restrict__ v,
                                const float* __restrict__ lam,
                                float* __restrict__ disp, int M) {
  const int b = blockIdx.y;
  const int i = blockIdx.x * 256 + threadIdx.x;
  if (i < M) disp[(size_t)b * M + i] = 3.0f * lam[b] * v[(size_t)b * M + i];
}

// ----------------------------- launcher ------------------------------------
extern "C" void kernel_launch(void* const* d_in, const int* in_sizes, int n_in,
                              void* d_out, int out_size, void* d_ws, size_t ws_size,
                              hipStream_t stream) {
  const float* coords = (const float*)d_in[0];
  const int* num_atoms = (const int*)d_in[1];
  const int B = in_sizes[1];
  const int M = in_sizes[0] / B;  // 3N = 3072 (divisible by 128)
  const int N = M / 3;
  const size_t BM = (size_t)B * M;
  const size_t BMM = BM * (size_t)M;

  float* H = (float*)d_out;                       // [B, M, M] fp32
  float* disp = H + BMM;                          // [B, M]

  // workspace: optional bf16 H mirror first, then small vectors
  const size_t mirror_bytes = BMM * sizeof(bf16_t);
  const size_t small_bytes = BM * (4 + 4 + 2 + 2) + (size_t)B * 4 + 256;
  const bool big = (ws_size >= mirror_bytes + small_bytes);

  char* wp = (char*)d_ws;
  bf16_t* H16 = nullptr;
  if (big) { H16 = (bf16_t*)wp; wp += mirror_bytes; }
  float* v = (float*)wp;            wp += BM * 4;
  float* w = (float*)wp;            wp += BM * 4;
  bf16_t* v16 = (bf16_t*)wp;        wp += BM * 2;
  bf16_t* w16 = (bf16_t*)wp;        wp += BM * 2;
  float* lam = (float*)wp;

  build_hessian_kernel<<<dim3(N, B), 256, 0, stream>>>(coords, num_atoms, H, H16, N, M);

  const int totalBM = (int)BM;
  init_v_kernel<<<(totalBM + 255) / 256, 256, 0, stream>>>(v, v16, totalBM);

  const dim3 gw(M / 16, B);
  if (big) {
    // 22 groups of 3 unnormalized bf16 WMMA matvecs (range-safe: |w| <= l^3),
    // normalize once per group. Iterate lives in (v, v16) at group boundaries.
    const int GROUPS = 22;
    for (int g = 0; g < GROUPS; ++g) {
      matvec_wmma_bf16_kernel<<<gw, 128, 0, stream>>>(H16, v16, w, w16, M);  // w = H v
      matvec_wmma_bf16_kernel<<<gw, 128, 0, stream>>>(H16, w16, v, v16, M);  // v = H w
      matvec_wmma_bf16_kernel<<<gw, 128, 0, stream>>>(H16, v16, w, w16, M);  // w = H v
      normalize_kernel<<<B, 256, 0, stream>>>(v, w, lam, v16, M);
    }
  } else {
    const int NB_ITERS = 66;
    for (int it = 0; it < NB_ITERS; ++it) {
      matvec_wmma_cvt_kernel<<<gw, 128, 0, stream>>>(H, v, w, M);
      normalize_kernel<<<B, 256, 0, stream>>>(v, w, lam, v16, M);
    }
  }

  // fp32 polish + Rayleigh lambda
  const dim3 gf(M / 8, B);
  for (int it = 0; it < 4; ++it) {
    matvec_f32_kernel<<<gf, 256, 0, stream>>>(H, v, w, M);
    normalize_kernel<<<B, 256, 0, stream>>>(v, w, lam, v16, M);
  }
  finalize_kernel<<<dim3((M + 255) / 256, B), 256, 0, stream>>>(v, lam, disp, M);
}
